// GUET_5025111736964
// MI455X (gfx1250) — matrix operand order repaired and checked by reading, not compiled, not run on hardware
//
#include <hip/hip_runtime.h>
#include <math.h>

// ---------------- problem constants (match reference) ----------------
#define N_MI 10000
#define N_DI 5000
#define DD   64
#define E_MI 320000
#define E_DI 160000
#define PP   200000
#define NK   8                      // GRID_SIZE + SPLINE_K
#define WMI  ((N_DI + 31) / 32)     // occupancy words per pay_mi row (157)
#define WDI  ((N_MI + 31) / 32)     // occupancy words per pay_di row (313)

typedef __attribute__((ext_vector_type(16))) _Float16 v16h;
typedef __attribute__((ext_vector_type(8)))  float    v8f;

__device__ __forceinline__ float fast_sigmoid(float x) {
    return __builtin_amdgcn_rcpf(1.f + __expf(-x));     // v_rcp_f32, no IEEE div chain
}
__device__ __forceinline__ float fast_silu(float x) { return x * fast_sigmoid(x); }

// =====================================================================
// WMMA GEMM:  C[M,N] = act(A[gidx[m],:K]) @ W[N,K]^T (+ bias)
// One wave per 16x16 tile; K (64/32) and N (64/32/16) compile-time.
// Per lane, fragment data is CONTIGUOUS in memory:
//   A: [kb+8*hi .. +7] and [kb+16+8*hi .. +7]   -> 4x float4 loads
//   B: W[n, kb+8*hi .. +7] and [kb+16+8*hi .. +7] -> 4x float4 loads
// Out-of-range M lanes clamp their row index (their C rows are never
// stored, so no zero-masking / exec manipulation is needed). N is a
// multiple of 16 with exact grids -> no column checks at all.
// =====================================================================
template <int K, int N, bool SILU>
__global__ __launch_bounds__(32)
void k_wmma_gemm(const float* __restrict__ A, const int* __restrict__ gidx,
                 const float* __restrict__ W, const float* __restrict__ bias,
                 float* __restrict__ C, int M)
{
    static_assert(N % 16 == 0, "N tile-exact");
    const int lane = threadIdx.x;      // 0..31
    const int hi   = lane >> 4;
    const int ln   = lane & 15;
    const int m0   = blockIdx.x * 16;
    const int n    = blockIdx.y * 16 + ln;           // always < N
    int m = m0 + ln; if (m >= M) m = M - 1;          // clamp (safe: row-private)

    const int r = gidx ? gidx[m] : m;
    const float* __restrict__ arow = A + (size_t)r * K;
    const float* __restrict__ wrow = W + (size_t)n * K;
    if (gidx) __builtin_prefetch(arow, 0, 0);        // global_prefetch_b8 (gathered rows)

    v8f acc = {0.f, 0.f, 0.f, 0.f, 0.f, 0.f, 0.f, 0.f};
#pragma unroll
    for (int kb = 0; kb < K; kb += 32) {
        const float4* alo = (const float4*)(arow + kb + hi * 8);
        const float4* ahi = (const float4*)(arow + kb + 16 + hi * 8);
        const float4* bpv = (const float4*)(wrow + kb + hi * 8);
        const float4* bph = (const float4*)(wrow + kb + 16 + hi * 8);
        float4 a0 = alo[0], a1 = alo[1], a2 = ahi[0], a3 = ahi[1];
        float4 b0 = bpv[0], b1 = bpv[1], b2 = bph[0], b3 = bph[1];
        float af[16] = {a0.x, a0.y, a0.z, a0.w, a1.x, a1.y, a1.z, a1.w,
                        a2.x, a2.y, a2.z, a2.w, a3.x, a3.y, a3.z, a3.w};
        float bf[16] = {b0.x, b0.y, b0.z, b0.w, b1.x, b1.y, b1.z, b1.w,
                        b2.x, b2.y, b2.z, b2.w, b3.x, b3.y, b3.z, b3.w};
        v16h a, b;
#pragma unroll
        for (int j = 0; j < 16; ++j) {
            float av = af[j];
            if (SILU) av = fast_silu(av);
            a[j] = (_Float16)av;
            b[j] = (_Float16)bf[j];
        }
        acc = __builtin_amdgcn_wmma_f32_16x16x32_f16(
            false, a, false, b, (short)0, acc, false, false);
    }
    float bv = 0.f;
    if (bias) bv = bias[n];
    // C/D layout: VGPR j holds row m0 + j + 8*hi, column n
    float* __restrict__ crow = C + (size_t)(m0 + 8 * hi) * N + n;
    if (m0 + 16 <= M) {                 // uniform fast path: no exec masking
#pragma unroll
        for (int j = 0; j < 8; ++j) crow[j * N] = acc[j] + bv;
    } else {                            // M tail (di graph only)
#pragma unroll
        for (int j = 0; j < 8; ++j)
            if (m0 + j + 8 * hi < M) crow[j * N] = acc[j] + bv;
    }
}

// ---------------- small utility kernels ----------------
__global__ void k_fill_u32(unsigned int* p, unsigned int v, int n) {
    int i = blockIdx.x * blockDim.x + threadIdx.x;
    if (i < n) p[i] = v;
}
__global__ void k_set_f32(float* p, float v, int n) {
    int i = blockIdx.x * blockDim.x + threadIdx.x;
    if (i < n) p[i] = v;
}
__global__ void k_avg_inplace(float* a, const float* b, int n) {
    int i = blockIdx.x * blockDim.x + threadIdx.x;
    if (i < n) a[i] = 0.5f * (a[i] + b[i]);
}
__global__ void k_mul_inplace(float* a, const float* b, int n) {
    int i = blockIdx.x * blockDim.x + threadIdx.x;
    if (i < n) a[i] *= b[i];
}

// ---------------- GCN graph kernels ----------------
__global__ void k_deg_edges(const int* __restrict__ dst, const float* __restrict__ ew,
                            float* deg, int E) {
    int e = blockIdx.x * blockDim.x + threadIdx.x;
    if (e < E) atomicAdd(&deg[dst[e]], ew[e]);
}
__global__ void k_dinv(float* deg, int n) {
    int i = blockIdx.x * blockDim.x + threadIdx.x;
    if (i < n) { float d = deg[i]; deg[i] = (d > 0.f) ? rsqrtf(d) : 0.f; }
}
// self-loop contribution: agg[i,c] = xw[i,c] * dinv[i]^2  (also zero-initializes agg)
__global__ void k_selfloop(const float* __restrict__ xw, const float* __restrict__ dinv,
                           float* __restrict__ agg, int n) {
    int i = blockIdx.x * blockDim.x + threadIdx.x;
    if (i < n * DD) {
        float dv = dinv[i >> 6];
        agg[i] = xw[i] * dv * dv;
    }
}
__global__ void k_scatter(const int* __restrict__ src, const int* __restrict__ dst,
                          const float* __restrict__ ew, const float* __restrict__ dinv,
                          const float* __restrict__ xw, float* __restrict__ agg, int E) {
    int i = blockIdx.x * blockDim.x + threadIdx.x;
    if (i >= E * DD) return;
    int e = i >> 6, c = i & 63;
    int s = src[e], d = dst[e];
    float coef = dinv[s] * ew[e] * dinv[d];
    atomicAdd(&agg[d * DD + c], xw[s * DD + c] * coef);
}
__global__ void k_bias_relu(const float* __restrict__ agg, const float* __restrict__ b,
                            float* __restrict__ out, int n) {
    int i = blockIdx.x * blockDim.x + threadIdx.x;
    if (i < n * DD) {
        float v = agg[i] + b[i & 63];
        out[i] = v > 0.f ? v : 0.f;
    }
}

// ---------------- game theory ----------------
__global__ __launch_bounds__(256)
void k_rewards(const float* __restrict__ a, const float* __restrict__ b,
               float* __restrict__ r, int P) {
    int p = blockIdx.x * blockDim.x + threadIdx.x;
    if (p >= P) return;
    const float* ap = a + (size_t)p * DD;
    const float* bp = b + (size_t)p * DD;
    float dot = 0.f, na = 0.f, nb = 0.f;
#pragma unroll
    for (int c = 0; c < DD; ++c) {
        float x = ap[c], y = bp[c];
        dot += x * y; na += x * x; nb += y * y;
    }
    r[p] = dot * rsqrtf(na) * rsqrtf(nb);
}

__device__ __forceinline__ unsigned int ordf(float f) {
    unsigned int u = __float_as_uint(f);
    return (u & 0x80000000u) ? ~u : (u | 0x80000000u);   // monotone float->uint
}

// scatter rewards into per-row packed max keys + occupancy bitmaps
__global__ void k_payoff(const int* __restrict__ mi, const int* __restrict__ di,
                         const float* __restrict__ rw,
                         unsigned long long* pack_mi, unsigned long long* pack_di,
                         unsigned int* occ_mi, unsigned int* occ_di, int P) {
    int p = blockIdx.x * blockDim.x + threadIdx.x;
    if (p >= P) return;
    int r = mi[p], c = di[p];
    unsigned long long hikey = ((unsigned long long)ordf(rw[p])) << 32;
    atomicMax(&pack_mi[r], hikey | (unsigned int)(~(unsigned int)c)); // smaller col wins ties
    atomicMax(&pack_di[c], hikey | (unsigned int)(~(unsigned int)r));
    atomicOr(&occ_mi[(size_t)r * WMI + (c >> 5)], 1u << (c & 31));
    atomicOr(&occ_di[(size_t)c * WDI + (r >> 5)], 1u << (r & 31));
}

// argmax over a dense row of zeros with scattered rewards (jnp.argmax semantics)
__global__ void k_argmax(const unsigned long long* __restrict__ pack,
                         const unsigned int* __restrict__ occ,
                         int* __restrict__ bcol, int rows, int cols, int words) {
    int r = blockIdx.x * blockDim.x + threadIdx.x;
    if (r >= rows) return;
    int z = -1;                                  // first column with implicit zero
    const unsigned int* row = occ + (size_t)r * words;
    for (int w = 0; w < words; ++w) {
        unsigned int mask = 0xFFFFFFFFu;
        int rem = cols - w * 32;
        if (rem < 32) mask = (rem <= 0) ? 0u : ((1u << rem) - 1u);
        unsigned int freeb = (~row[w]) & mask;
        if (freeb) { z = w * 32 + (__ffs(freeb) - 1); break; }
    }
    unsigned long long k = pack[r];
    int best;
    if (k == 0ULL) {                             // no pairs in this row: all zeros
        best = (z >= 0) ? z : 0;
    } else {
        unsigned int ou  = (unsigned int)(k >> 32);
        unsigned int raw = (ou & 0x80000000u) ? (ou ^ 0x80000000u) : ~ou;
        float rv = __uint_as_float(raw);
        int col = (int)(~((unsigned int)(k & 0xFFFFFFFFull)));
        if (rv > 0.f)       best = col;
        else if (rv == 0.f) best = (z >= 0 && z < col) ? z : col;
        else                best = (z >= 0) ? z : col;   // zeros beat negative max
    }
    bcol[r] = best;
}

// best-strategy gather + nash partial + 0.5/0.25/0.25 mix + LayerNorm
__global__ __launch_bounds__(256)
void k_finalize(const float* __restrict__ e, const int* __restrict__ index,
                const int* __restrict__ bcol, const float* __restrict__ nmf,
                const float* __restrict__ svd, const float* __restrict__ g,
                const float* __restrict__ bta, float* __restrict__ outf,
                float* __restrict__ nash_acc, int P) {
    int p = blockIdx.x * blockDim.x + threadIdx.x;
    float sq = 0.f;
    if (p < P) {
        int idx = index[p];
        int brow = bcol[idx];                     // < N_DI (or N_MI) <= P rows of e
        const float* ep = e + (size_t)p * DD;
        const float* bp = e + (size_t)brow * DD;
        const float* np = nmf + (size_t)idx * DD;
        const float* sp = svd + (size_t)idx * DD;
        float v[DD];
        float mean = 0.f;
#pragma unroll
        for (int c = 0; c < DD; ++c) {
            float bs = bp[c];
            float d = ep[c] - bs;
            sq += d * d;
            float val = 0.5f * bs + 0.25f * np[c] + 0.25f * sp[c];
            v[c] = val; mean += val;
        }
        mean *= (1.f / DD);
        float var = 0.f;
#pragma unroll
        for (int c = 0; c < DD; ++c) { float d = v[c] - mean; var += d * d; }
        var *= (1.f / DD);
        float inv = rsqrtf(var + 1e-5f);
        float* op = outf + (size_t)p * DD;
#pragma unroll
        for (int c = 0; c < DD; ++c) op[c] = (v[c] - mean) * inv * g[c] + bta[c];
    }
    __shared__ float red[256];
    red[threadIdx.x] = sq;
    __syncthreads();
    for (int s = 128; s > 0; s >>= 1) {
        if ((int)threadIdx.x < s) red[threadIdx.x] += red[threadIdx.x + s];
        __syncthreads();
    }
    if (threadIdx.x == 0) atomicAdd(nash_acc, red[0]);
}

// ---------------- KAN (B-splines, degree 3, uniform grid g(i)=0.4(i-3)-1) ----
__device__ __forceinline__ void bspline8(float x, float* b8) {
    float b[11];
#pragma unroll
    for (int j = 0; j < 11; ++j) {
        float gj  = 0.4f * (float)(j - 3) - 1.0f;
        float gj1 = 0.4f * (float)(j - 2) - 1.0f;
        b[j] = (x >= gj && x < gj1) ? 1.f : 0.f;
    }
#pragma unroll
    for (int p = 1; p <= 3; ++p) {
        float inv = 2.5f / (float)p;              // 1 / (p * 0.4)
#pragma unroll
        for (int j = 0; j < 11 - p; ++j) {
            float gj   = 0.4f * (float)(j - 3) - 1.0f;       // grid[j]
            float gjp1 = 0.4f * (float)(j + p - 2) - 1.0f;   // grid[j+p+1]
            b[j] = (x - gj) * inv * b[j] + (gjp1 - x) * inv * b[j + 1];
        }
    }
#pragma unroll
    for (int k = 0; k < 8; ++k) b8[k] = b[k];
}

// out[o,i,k] = spline_w[o,i,k] * scaler[o,i]
__global__ void k_sw_combine(const float* __restrict__ sw, const float* __restrict__ sc,
                             float* __restrict__ out, int outin) {
    int i = blockIdx.x * blockDim.x + threadIdx.x;
    if (i < outin * NK) out[i] = sw[i] * sc[i / NK];
}

// out[p,o] += sum_{i,k} B(x[p,i])_k * SW[o,i,k]; SW staged in LDS
template <int IN, int OUT>
__global__ __launch_bounds__(256)
void k_kan_spline(const float* __restrict__ x, const float* __restrict__ SW,
                  float* __restrict__ out, int P) {
    extern __shared__ float sw[];                 // OUT*IN*NK floats
    for (int i = threadIdx.x; i < OUT * IN * NK; i += blockDim.x) sw[i] = SW[i];
    __syncthreads();
    int p = blockIdx.x * blockDim.x + threadIdx.x;
    if (p >= P) return;
    float acc[OUT];
#pragma unroll
    for (int o = 0; o < OUT; ++o) acc[o] = 0.f;
    for (int i = 0; i < IN; ++i) {
        float xv = x[(size_t)p * IN + i];
        float bas[8];
        bspline8(xv, bas);
        for (int o = 0; o < OUT; ++o) {
            const float4* wp = (const float4*)&sw[(o * IN + i) * NK];  // 32B aligned
            float4 w0 = wp[0], w1 = wp[1];                              // 2x ds_load_b128
            float s = bas[0] * w0.x + bas[1] * w0.y + bas[2] * w0.z + bas[3] * w0.w +
                      bas[4] * w1.x + bas[5] * w1.y + bas[6] * w1.z + bas[7] * w1.w;
            acc[o] += s;
        }
    }
    float* op = out + (size_t)p * OUT;
#pragma unroll
    for (int o = 0; o < OUT; ++o) op[o] += acc[o];
}

// final KAN layer (16 -> 1) + sigmoid
__global__ __launch_bounds__(256)
void k_kan3(const float* __restrict__ pfeat, const float* __restrict__ bw,
            const float* __restrict__ sw3, float* __restrict__ scores, int P) {
    int p = blockIdx.x * blockDim.x + threadIdx.x;
    if (p >= P) return;
    float s = 0.f;
#pragma unroll
    for (int i = 0; i < 16; ++i) {
        float x = pfeat[(size_t)p * 16 + i];
        s += fast_silu(x) * bw[i];
        float bas[8];
        bspline8(x, bas);
#pragma unroll
        for (int k = 0; k < 8; ++k) s += bas[k] * sw3[i * 8 + k];
    }
    scores[p] = fast_sigmoid(s);
}

__global__ void k_nash(const float* acc, float* out) {
    out[0] = acc[0] / (2.f * (float)PP * (float)DD);
}

// =====================================================================
// host orchestration
// =====================================================================
extern "C" void kernel_launch(void* const* d_in, const int* in_sizes, int n_in,
                              void* d_out, int out_size, void* d_ws, size_t ws_size,
                              hipStream_t stream)
{
    (void)in_sizes; (void)n_in; (void)out_size; (void)ws_size;
    // inputs in setup_inputs() order; params dict flattened in insertion order
    const float* mi_infeats = (const float*)d_in[0];
    const float* di_infeats = (const float*)d_in[1];
    const float* mi_edge_w  = (const float*)d_in[2];
    const float* di_edge_w  = (const float*)d_in[3];
    const float* mi_svd     = (const float*)d_in[4];
    const float* mi_nmf     = (const float*)d_in[5];
    const float* di_svd     = (const float*)d_in[6];
    const float* di_nmf     = (const float*)d_in[7];
    const float* mi_W1 = (const float*)d_in[8];
    const float* mi_b1 = (const float*)d_in[9];
    const float* mi_W2 = (const float*)d_in[10];
    const float* mi_b2 = (const float*)d_in[11];
    const float* di_W1 = (const float*)d_in[12];
    const float* di_b1 = (const float*)d_in[13];
    const float* di_W2 = (const float*)d_in[14];
    const float* di_b2 = (const float*)d_in[15];
    const float* gt_mi_W = (const float*)d_in[16];
    const float* gt_mi_b = (const float*)d_in[17];
    const float* gt_di_W = (const float*)d_in[18];
    const float* gt_di_b = (const float*)d_in[19];
    const float* ln_g = (const float*)d_in[20];
    const float* ln_b = (const float*)d_in[21];
    const float* kan1_base   = (const float*)d_in[22];
    const float* kan1_spline = (const float*)d_in[23];
    const float* kan1_scaler = (const float*)d_in[24];
    const float* kan2_base   = (const float*)d_in[25];
    const float* kan2_spline = (const float*)d_in[26];
    const float* kan2_scaler = (const float*)d_in[27];
    const float* kan3_base   = (const float*)d_in[28];
    const float* kan3_spline = (const float*)d_in[29];
    const float* kan3_scaler = (const float*)d_in[30];
    const int* mi_edges = (const int*)d_in[31];
    const int* di_edges = (const int*)d_in[32];
    const int* mi_index = (const int*)d_in[33];
    const int* di_index = (const int*)d_in[34];

    const int* mi_src = mi_edges;           const int* mi_dst = mi_edges + E_MI;
    const int* di_src = di_edges;           const int* di_dst = di_edges + E_DI;

    // ---- workspace carve-out (total ~270 MB) ----
    char* wsb = (char*)d_ws;
    size_t off = 0;
    auto carve = [&](size_t bytes) -> void* {
        void* p = wsb + off;
        off += (bytes + 255) & ~(size_t)255;
        return p;
    };
    float* deg_mi  = (float*)carve(N_MI * 4);
    float* deg_di  = (float*)carve(N_DI * 4);
    float* xw_mi   = (float*)carve((size_t)N_MI * DD * 4);
    float* agg_mi  = (float*)carve((size_t)N_MI * DD * 4);
    float* m1_mi   = (float*)carve((size_t)N_MI * DD * 4);
    float* m2_mi   = (float*)carve((size_t)N_MI * DD * 4);
    float* xw_di   = (float*)carve((size_t)N_DI * DD * 4);
    float* agg_di  = (float*)carve((size_t)N_DI * DD * 4);
    float* m1_di   = (float*)carve((size_t)N_DI * DD * 4);
    float* m2_di   = (float*)carve((size_t)N_DI * DD * 4);
    float* mi_e    = (float*)carve((size_t)PP * DD * 4);
    float* di_e    = (float*)carve((size_t)PP * DD * 4);
    float* rewards = (float*)carve((size_t)PP * 4);
    unsigned long long* pack_mi = (unsigned long long*)carve(N_MI * 8);
    unsigned long long* pack_di = (unsigned long long*)carve(N_DI * 8);
    unsigned int* occ_mi = (unsigned int*)carve((size_t)N_MI * WMI * 4);
    unsigned int* occ_di = (unsigned int*)carve((size_t)N_DI * WDI * 4);
    int* bcol_mi = (int*)carve(N_MI * 4);
    int* bcol_di = (int*)carve(N_DI * 4);
    float* nash_acc = (float*)carve(4);
    float* mi_final = (float*)carve((size_t)PP * DD * 4);
    float* di_final = (float*)carve((size_t)PP * DD * 4);
    float* kout1    = (float*)carve((size_t)PP * 32 * 4);
    float* SW1 = (float*)carve(32 * 64 * NK * 4);
    float* SW2 = (float*)carve(16 * 32 * NK * 4);
    float* SW3 = (float*)carve(16 * NK * 4);

    auto nb = [](long long n, int b) { return dim3((unsigned)((n + b - 1) / b)); };
    // D=64 output GEMM (gcn / game-theory layers)
    auto gemm64x64 = [&](const float* A, const int* gi, const float* W, const float* bias,
                         float* C, int M) {
        dim3 grid((M + 15) / 16, DD / 16);
        k_wmma_gemm<64, 64, false><<<grid, dim3(32), 0, stream>>>(A, gi, W, bias, C, M);
    };
    auto gcn_layer = [&](const float* x, const float* W, const float* b,
                         const int* src, const int* dst, const float* ew,
                         const float* dinv, float* xw, float* agg, float* out,
                         int n, int E) {
        gemm64x64(x, nullptr, W, nullptr, xw, n);
        k_selfloop<<<nb((long long)n * DD, 256), 256, 0, stream>>>(xw, dinv, agg, n);
        k_scatter<<<nb((long long)E * DD, 256), 256, 0, stream>>>(src, dst, ew, dinv, xw, agg, E);
        k_bias_relu<<<nb((long long)n * DD, 256), 256, 0, stream>>>(agg, b, out, n);
    };

    // ---- GCN degree / normalization (self-loop weight 1 folded into init) ----
    k_set_f32<<<nb(N_MI, 256), 256, 0, stream>>>(deg_mi, 1.0f, N_MI);
    k_deg_edges<<<nb(E_MI, 256), 256, 0, stream>>>(mi_dst, mi_edge_w, deg_mi, E_MI);
    k_dinv<<<nb(N_MI, 256), 256, 0, stream>>>(deg_mi, N_MI);
    k_set_f32<<<nb(N_DI, 256), 256, 0, stream>>>(deg_di, 1.0f, N_DI);
    k_deg_edges<<<nb(E_DI, 256), 256, 0, stream>>>(di_dst, di_edge_w, deg_di, E_DI);
    k_dinv<<<nb(N_DI, 256), 256, 0, stream>>>(deg_di, N_DI);

    // ---- two GCN layers per graph, feat = 0.5*(m1+m2) (into m1 buffer) ----
    gcn_layer(mi_infeats, mi_W1, mi_b1, mi_src, mi_dst, mi_edge_w, deg_mi,
              xw_mi, agg_mi, m1_mi, N_MI, E_MI);
    gcn_layer(m1_mi, mi_W2, mi_b2, mi_src, mi_dst, mi_edge_w, deg_mi,
              xw_mi, agg_mi, m2_mi, N_MI, E_MI);
    k_avg_inplace<<<nb((long long)N_MI * DD, 256), 256, 0, stream>>>(m1_mi, m2_mi, N_MI * DD);

    gcn_layer(di_infeats, di_W1, di_b1, di_src, di_dst, di_edge_w, deg_di,
              xw_di, agg_di, m1_di, N_DI, E_DI);
    gcn_layer(m1_di, di_W2, di_b2, di_src, di_dst, di_edge_w, deg_di,
              xw_di, agg_di, m2_di, N_DI, E_DI);
    k_avg_inplace<<<nb((long long)N_DI * DD, 256), 256, 0, stream>>>(m1_di, m2_di, N_DI * DD);

    // ---- game theory embeddings (gather fused into GEMM A-load) ----
    gemm64x64(m1_mi, mi_index, gt_mi_W, gt_mi_b, mi_e, PP);
    gemm64x64(m1_di, di_index, gt_di_W, gt_di_b, di_e, PP);
    k_rewards<<<nb(PP, 256), 256, 0, stream>>>(mi_e, di_e, rewards, PP);

    // ---- payoff argmax without materializing [N_MI,N_DI] ----
    k_fill_u32<<<nb(N_MI * 2, 256), 256, 0, stream>>>((unsigned int*)pack_mi, 0u, N_MI * 2);
    k_fill_u32<<<nb(N_DI * 2, 256), 256, 0, stream>>>((unsigned int*)pack_di, 0u, N_DI * 2);
    k_fill_u32<<<nb((long long)N_MI * WMI, 256), 256, 0, stream>>>(occ_mi, 0u, N_MI * WMI);
    k_fill_u32<<<nb((long long)N_DI * WDI, 256), 256, 0, stream>>>(occ_di, 0u, N_DI * WDI);
    k_fill_u32<<<dim3(1), 32, 0, stream>>>((unsigned int*)nash_acc, 0u, 1);
    k_payoff<<<nb(PP, 256), 256, 0, stream>>>(mi_index, di_index, rewards,
                                              pack_mi, pack_di, occ_mi, occ_di, PP);
    k_argmax<<<nb(N_MI, 256), 256, 0, stream>>>(pack_mi, occ_mi, bcol_mi, N_MI, N_DI, WMI);
    k_argmax<<<nb(N_DI, 256), 256, 0, stream>>>(pack_di, occ_di, bcol_di, N_DI, N_MI, WDI);

    // ---- strategies + nash + mix + LayerNorm ----
    k_finalize<<<nb(PP, 256), 256, 0, stream>>>(mi_e, mi_index, bcol_mi, mi_nmf, mi_svd,
                                                ln_g, ln_b, mi_final, nash_acc, PP);
    k_finalize<<<nb(PP, 256), 256, 0, stream>>>(di_e, di_index, bcol_di, di_nmf, di_svd,
                                                ln_g, ln_b, di_final, nash_acc, PP);
    // h = mi_final * di_final (in place in mi_final)
    k_mul_inplace<<<nb((long long)PP * DD, 256), 256, 0, stream>>>(mi_final, di_final, PP * DD);

    // ---- KAN head ----
    k_sw_combine<<<nb(32 * 64 * NK, 256), 256, 0, stream>>>(kan1_spline, kan1_scaler, SW1, 32 * 64);
    k_sw_combine<<<nb(16 * 32 * NK, 256), 256, 0, stream>>>(kan2_spline, kan2_scaler, SW2, 16 * 32);
    k_sw_combine<<<nb(16 * NK, 256), 256, 0, stream>>>(kan3_spline, kan3_scaler, SW3, 16);

    float* scores = (float*)d_out;            // [P]
    float* nash   = scores + PP;              // [1]
    float* pfeat  = nash + 1;                 // [P,16]

    {   // kan1 base: K=64 -> N=32, silu on A
        dim3 grid((PP + 15) / 16, 32 / 16);
        k_wmma_gemm<64, 32, true><<<grid, dim3(32), 0, stream>>>(mi_final, nullptr, kan1_base,
                                                                 nullptr, kout1, PP);
    }
    k_kan_spline<64, 32><<<nb(PP, 256), 256, 32 * 64 * NK * 4, stream>>>(mi_final, SW1, kout1, PP);

    {   // kan2 base: K=32 -> N=16, silu on A
        dim3 grid((PP + 15) / 16, 1);
        k_wmma_gemm<32, 16, true><<<grid, dim3(32), 0, stream>>>(kout1, nullptr, kan2_base,
                                                                 nullptr, pfeat, PP);
    }
    k_kan_spline<32, 16><<<nb(PP, 256), 256, 16 * 32 * NK * 4, stream>>>(kout1, SW2, pfeat, PP);

    k_kan3<<<nb(PP, 256), 256, 0, stream>>>(pfeat, kan3_base, SW3, scores, PP);
    k_nash<<<dim3(1), 32, 0, stream>>>(nash_acc, nash);
}
